// PTA_EGNN_83236466196761
// MI455X (gfx1250) — compile-verified, matrix operand order checked by reading, not run on hardware
//
#include <hip/hip_runtime.h>
#include <math.h>

#define N_NODES 10000
#define N_EDGES 160000
#define MAX_K 8
#define HID 64
#define HEADS 4

typedef __attribute__((ext_vector_type(16))) __bf16 v16bf;
typedef __attribute__((ext_vector_type(8)))  float  v8f;
typedef __attribute__((ext_vector_type(4)))  unsigned int v4u;

union BfVec { v16bf v; v4u q[2]; unsigned short u[16]; };

__device__ __forceinline__ unsigned short f2bf(float f) {
    union { float f; unsigned u; } x; x.f = f;
    unsigned u = x.u + 0x7FFFu + ((x.u >> 16) & 1u);
    return (unsigned short)(u >> 16);
}
__device__ __forceinline__ float bflo(unsigned w) { return __uint_as_float(w << 16); }
__device__ __forceinline__ float bfhi(unsigned w) { return __uint_as_float(w & 0xffff0000u); }
__device__ __forceinline__ float sigm(float x) { return 1.0f / (1.0f + __expf(-x)); }
__device__ __forceinline__ float silu_(float x) { return x * sigm(x); }

__device__ __forceinline__ v8f v8zero() {
    v8f v;
#pragma unroll
    for (int i = 0; i < 8; i++) v[i] = 0.0f;
    return v;
}

// ---------- fragment-order swizzle for one 16x32 bf16 chunk (512 u16) ----------
__device__ __forceinline__ unsigned swz(unsigned m, unsigned k) {
    return (m + 16u * ((k >> 3) & 1u)) * 16u + ((k & 7u) + 8u * ((k >> 4) & 1u));
}
__device__ __forceinline__ unsigned swz64(unsigned m, unsigned j) {
    return (j >> 5) * 512u + swz(m, j & 31u);
}
__device__ __forceinline__ BfVec frag_ld(const unsigned short* p) {
    BfVec r;
    const v4u* q = (const v4u*)p;
    r.q[0] = q[0]; r.q[1] = q[1];
    return r;
}
__device__ __forceinline__ v8f wmma_bf(const BfVec& a, const BfVec& b, v8f c) {
    return __builtin_amdgcn_wmma_f32_16x16x32_bf16(false, a.v, false, b.v, (short)0, c, false, false);
}

// stage one 16x32 A chunk into fragment order; lane handles one 16-wide row half
__device__ __forceinline__ void stage_a_bf(unsigned short* dst, const unsigned short* src,
                                           unsigned lds, unsigned k0, unsigned K, int lane) {
    unsigned m = (unsigned)lane >> 1, kh = ((unsigned)lane & 1u) * 16u;
#pragma unroll
    for (unsigned t = 0; t < 16; t++) {
        unsigned k = k0 + kh + t;
        dst[swz(m, kh + t)] = (k < K) ? src[m * lds + k] : (unsigned short)0;
    }
}
__device__ __forceinline__ void stage_a_f32(unsigned short* dst, const float* src,
                                            unsigned lds, unsigned k0, unsigned K, int lane) {
    unsigned m = (unsigned)lane >> 1, kh = ((unsigned)lane & 1u) * 16u;
#pragma unroll
    for (unsigned t = 0; t < 16; t++) {
        unsigned k = k0 + kh + t;
        dst[swz(m, kh + t)] = (k < K) ? f2bf(src[m * lds + k]) : (unsigned short)0;
    }
}

// ---------------- weight convert + pre-swizzle: (Ksrc x 64) f32 -> fragment-ordered bf16 ----------------
__global__ void cvt_swz_kernel(const float* __restrict__ s, unsigned short* __restrict__ d,
                               int Ksrc, int Kpad) {
    int i = blockIdx.x * 256 + threadIdx.x;
    if (i >= Kpad * 64) return;
    unsigned k = (unsigned)i >> 6, c = (unsigned)i & 63u;
    unsigned short v = (k < (unsigned)Ksrc) ? f2bf(s[k * 64u + c]) : (unsigned short)0;
    d[(k >> 5) * 2048u + (c >> 4) * 512u + swz(c & 15u, k & 31u)] = v;
}
__global__ void zero_kernel(float* p, int n) {
    int i = blockIdx.x * 256 + threadIdx.x;
    if (i < n) p[i] = 0.0f;
}

// ---------------- generic WMMA GEMM: (M x K) @ (K x 64), 4 waves x 16 rows per block ----------------
template <int ACT, bool ACCUM, bool IN_BF16, bool HAS_BIAS, bool RESID, bool OUT_BF, bool OUT_F>
__global__ void gemm64(const void* __restrict__ Aptr, int lda, int K,
                       const unsigned short* __restrict__ W,   // swizzled bf16
                       const float* __restrict__ bias,
                       const float* __restrict__ accb,
                       const float* __restrict__ resid,
                       float* __restrict__ outF, unsigned short* __restrict__ outB,
                       int oldd, int ooff, int Mtiles) {
    __shared__ __align__(16) unsigned short sA[4][512];
    __shared__ __align__(16) unsigned short sW[2048];
    const int lane = threadIdx.x & 31;
    const int wid = threadIdx.x >> 5;
    const int tile = blockIdx.x * 4 + wid;
    const bool active = tile < Mtiles;
    const unsigned m0 = active ? (unsigned)tile * 16u : 0u;
    v8f acc[4];
#pragma unroll
    for (int t = 0; t < 4; t++) acc[t] = v8zero();

    for (int k0 = 0; k0 < K; k0 += 32) {
        if (IN_BF16) stage_a_bf(sA[wid], (const unsigned short*)Aptr + (size_t)m0 * lda,
                                (unsigned)lda, (unsigned)k0, (unsigned)K, lane);
        else         stage_a_f32(sA[wid], (const float*)Aptr + (size_t)m0 * lda,
                                 (unsigned)lda, (unsigned)k0, (unsigned)K, lane);
        if (wid == 0) {
            const v4u* src = (const v4u*)(W + (size_t)(k0 >> 5) * 2048);
            v4u* dst = (v4u*)sW;
            for (int i = lane; i < 256; i += 32) dst[i] = src[i];
            if (k0 + 32 < K) __builtin_prefetch(W + (size_t)((k0 + 32) >> 5) * 2048, 0, 1);
        }
        __syncthreads();
        BfVec a = frag_ld(sA[wid] + lane * 16);
#pragma unroll
        for (int t = 0; t < 4; t++)
            acc[t] = wmma_bf(a, frag_ld(sW + t * 512 + lane * 16), acc[t]);
        __syncthreads();
    }
    if (!active) return;
    unsigned hf = (unsigned)lane >> 4, n = (unsigned)lane & 15u;
#pragma unroll
    for (unsigned t = 0; t < 4; t++) {
#pragma unroll
        for (unsigned r = 0; r < 8; r++) {
            unsigned M = m0 + r + 8u * hf, j = t * 16u + n;
            float v = acc[t][r];
            if (ACCUM) v += accb[M * 64u + j];
            if (HAS_BIAS) v += bias[j];
            if (ACT == 1) v = silu_(v);
            else if (ACT == 2) v = sigm(v);
            if (RESID) v += resid[M * 64u + j];
            if (OUT_F)  outF[M * (unsigned)oldd + (unsigned)ooff + j] = v;
            if (OUT_BF) outB[M * (unsigned)oldd + (unsigned)ooff + j] = f2bf(v);
        }
    }
}

// ---------------- fused per-edge kernel: chem/pos MLPs, z, gating, coord scalars ----------------
__global__ void edge_fused(
    const float* __restrict__ coord, const int* __restrict__ edges,
    const float* __restrict__ nvecs, const float* __restrict__ edge_attr,
    const float* __restrict__ P1, const float* __restrict__ P2,
    const unsigned short* __restrict__ wc2b, const unsigned short* __restrict__ wshb,
    const unsigned short* __restrict__ wp2b, const unsigned short* __restrict__ wu1b,
    const unsigned short* __restrict__ wx1b,
    const unsigned short* __restrict__ wc1eb, const unsigned short* __restrict__ wp1eb,
    const float* __restrict__ w_pos1,
    const float* __restrict__ b_chem1, const float* __restrict__ b_chem2,
    const float* __restrict__ b_sh, const float* __restrict__ b_pos1,
    const float* __restrict__ b_pos2, const float* __restrict__ bu1,
    const float* __restrict__ bx1,
    const float* __restrict__ w_att, const float* __restrict__ b_att,
    const float* __restrict__ wu2, const float* __restrict__ bu2,
    const float* __restrict__ wx2, const float* __restrict__ bx2,
    const float* __restrict__ wb,
    unsigned short* __restrict__ zbf, float* __restrict__ barr,
    float* __restrict__ xsum, float* __restrict__ cntp) {

    __shared__ __align__(16) unsigned short sWc2[4096], sWsh[4096], sWp2[4096];
    __shared__ __align__(16) unsigned short sWc1e[2048], sWp1e[2048];
    __shared__ float sWp1s[20 * 64];
    __shared__ float sB1[64], sB2[64], sBsh[64], sBp1[64], sBp2[64], sBu1[64], sBx1[64];
    __shared__ float sAtt[64], sWu2[64], sWx2[64];
    __shared__ __align__(16) unsigned short sA[1024];    // 16x64 fragment-ordered stage
    __shared__ __align__(16) unsigned short sChem[1024];
    __shared__ __align__(16) unsigned short sPos[1024];
    __shared__ float sScr[16 * 64];
    __shared__ int sRow[16], sCol[16];
    __shared__ float sCd[48];
    __shared__ float sRed[16];

    const int lane = threadIdx.x;
    const unsigned e0 = (unsigned)blockIdx.x * 16u;
    const unsigned hf = (unsigned)lane >> 4, nn = (unsigned)lane & 15u;

    for (int i = lane; i < 512; i += 32) {
        ((v4u*)sWc2)[i] = ((const v4u*)wc2b)[i];
        ((v4u*)sWsh)[i] = ((const v4u*)wshb)[i];
        ((v4u*)sWp2)[i] = ((const v4u*)wp2b)[i];
    }
    for (int i = lane; i < 256; i += 32) {
        ((v4u*)sWc1e)[i] = ((const v4u*)wc1eb)[i];
        ((v4u*)sWp1e)[i] = ((const v4u*)wp1eb)[i];
    }
    for (int i = lane; i < 1280; i += 32) sWp1s[i] = w_pos1[i];
    for (int i = lane; i < 64; i += 32) {
        sB1[i] = b_chem1[i]; sB2[i] = b_chem2[i]; sBsh[i] = b_sh[i]; sBp1[i] = b_pos1[i];
        sBp2[i] = b_pos2[i]; sBu1[i] = bu1[i]; sBx1[i] = bx1[i];
        sAtt[i] = w_att[i]; sWu2[i] = wu2[i]; sWx2[i] = wx2[i];
    }
    if (lane < 16) {
        unsigned e = e0 + (unsigned)lane;
        int r = edges[e], c = edges[N_EDGES + e];
        sRow[lane] = r; sCol[lane] = c;
        unsigned r3 = (unsigned)r * 3u, c3 = (unsigned)c * 3u;
        float dx = coord[r3] - coord[c3];
        float dy = coord[r3 + 1] - coord[c3 + 1];
        float dz = coord[r3 + 2] - coord[c3 + 2];
        float nrm = sqrtf(dx * dx + dy * dy + dz * dz) + 1e-8f;
        sCd[lane * 3] = dx / nrm; sCd[lane * 3 + 1] = dy / nrm; sCd[lane * 3 + 2] = dz / nrm;
    }
    __syncthreads();

    // ---- chem layer1: ea@Wc1e (WMMA, K=16 padded) + gathered P1/P2 + bias, silu -> sA
    stage_a_f32(sA, edge_attr + (size_t)e0 * 16, 16u, 0u, 16u, lane);
    __syncthreads();
    {
        BfVec a = frag_ld(sA + lane * 16);
        v8f acc[4];
#pragma unroll
        for (int t = 0; t < 4; t++)
            acc[t] = wmma_bf(a, frag_ld(sWc1e + t * 512 + lane * 16), v8zero());
        __syncthreads();
#pragma unroll
        for (unsigned t = 0; t < 4; t++)
#pragma unroll
            for (unsigned r = 0; r < 8; r++) {
                unsigned M = r + 8u * hf, j = t * 16u + nn;
                unsigned o1 = (unsigned)sRow[M] * 64u + j;
                unsigned o2 = (unsigned)sCol[M] * 64u + j;
                float v = acc[t][r] + P1[o1] + P2[o2] + sB1[j];
                sA[swz64(M, j)] = f2bf(silu_(v));
            }
        __syncthreads();
    }

    // ---- chem = silu(l1 @ Wc2 + b2) -> sChem
    {
        v8f acc[4];
#pragma unroll
        for (int t = 0; t < 4; t++) acc[t] = v8zero();
#pragma unroll
        for (int k0 = 0; k0 < 64; k0 += 32) {
            BfVec a = frag_ld(sA + (k0 >> 5) * 512 + lane * 16);
#pragma unroll
            for (int t = 0; t < 4; t++)
                acc[t] = wmma_bf(a, frag_ld(sWc2 + (k0 >> 5) * 2048 + t * 512 + lane * 16), acc[t]);
        }
#pragma unroll
        for (unsigned t = 0; t < 4; t++)
#pragma unroll
            for (unsigned r = 0; r < 8; r++) {
                unsigned M = r + 8u * hf, j = t * 16u + nn;
                sChem[swz64(M, j)] = f2bf(silu_(acc[t][r] + sB2[j]));
            }
        __syncthreads();
    }

    // ---- pos small features: nprod(5) + radial(15)
    if (lane < 16) {
        unsigned rI = (unsigned)sRow[lane] * 15u, cI = (unsigned)sCol[lane] * 15u;
#pragma unroll
        for (int v = 0; v < 5; v++) {
            float d = 0;
#pragma unroll
            for (int k = 0; k < 3; k++) d += nvecs[rI + v * 3 + k] * nvecs[cI + v * 3 + k];
            sScr[lane * 20 + v] = d;
        }
        float ir = 0;
#pragma unroll
        for (int k = 0; k < 3; k++) ir += sCd[lane * 3 + k] * sCd[lane * 3 + k];
        float sc = -0.5f;
#pragma unroll
        for (int k = 0; k < 15; k++) { sScr[lane * 20 + 5 + k] = __expf(ir * sc); sc *= (1.0f / 2.25f); }
    }
    __syncthreads();

    // ---- pos layer1: ea@Wp1e (WMMA) + small 20-dim dense + bias, silu -> sA
    stage_a_f32(sA, edge_attr + (size_t)e0 * 16, 16u, 0u, 16u, lane);
    __syncthreads();
    {
        BfVec a = frag_ld(sA + lane * 16);
        v8f acc[4];
#pragma unroll
        for (int t = 0; t < 4; t++)
            acc[t] = wmma_bf(a, frag_ld(sWp1e + t * 512 + lane * 16), v8zero());
        __syncthreads();
#pragma unroll
        for (unsigned t = 0; t < 4; t++)
#pragma unroll
            for (unsigned r = 0; r < 8; r++) {
                unsigned M = r + 8u * hf, j = t * 16u + nn;
                float v = acc[t][r] + sBp1[j];
                for (int u = 0; u < 20; u++) v += sScr[M * 20 + u] * sWp1s[u * 64 + j];
                sA[swz64(M, j)] = f2bf(silu_(v));
            }
        __syncthreads();
    }

    // ---- pos = silu(l1p @ Wp2 + bp2): regs + sPos
    v8f posv[4];
#pragma unroll
    for (int t = 0; t < 4; t++) posv[t] = v8zero();
#pragma unroll
    for (int k0 = 0; k0 < 64; k0 += 32) {
        BfVec a = frag_ld(sA + (k0 >> 5) * 512 + lane * 16);
#pragma unroll
        for (int t = 0; t < 4; t++)
            posv[t] = wmma_bf(a, frag_ld(sWp2 + (k0 >> 5) * 2048 + t * 512 + lane * 16), posv[t]);
    }
#pragma unroll
    for (unsigned t = 0; t < 4; t++)
#pragma unroll
        for (unsigned r = 0; r < 8; r++) {
            unsigned M = r + 8u * hf, j = t * 16u + nn;
            float v = silu_(posv[t][r] + sBp2[j]);
            posv[t][r] = v;
            sPos[swz64(M, j)] = f2bf(v);
        }
    __syncthreads();

    // ---- z_pre = silu(chem @ Wsh + bsh) * pos
    v8f zpv[4];
#pragma unroll
    for (int t = 0; t < 4; t++) zpv[t] = v8zero();
#pragma unroll
    for (int k0 = 0; k0 < 64; k0 += 32) {
        BfVec a = frag_ld(sChem + (k0 >> 5) * 512 + lane * 16);
#pragma unroll
        for (int t = 0; t < 4; t++)
            zpv[t] = wmma_bf(a, frag_ld(sWsh + (k0 >> 5) * 2048 + t * 512 + lane * 16), zpv[t]);
    }
#pragma unroll
    for (unsigned t = 0; t < 4; t++)
#pragma unroll
        for (unsigned r = 0; r < 8; r++) {
            unsigned M = r + 8u * hf, j = t * 16u + nn;
            float v = silu_(zpv[t][r] + sBsh[j]) * posv[t][r];
            zpv[t][r] = v;
            sScr[M * 64u + j] = v;
        }
    __syncthreads();

    // ---- attention gate scalar, z, per-head b dots
    if (lane < 16) {
        float s = b_att[0];
        for (int j = 0; j < 64; j++) s += sScr[lane * 64 + j] * sAtt[j];
        sRed[lane] = sigm(s);
    }
    __syncthreads();
#pragma unroll
    for (unsigned t = 0; t < 4; t++)
#pragma unroll
        for (unsigned r = 0; r < 8; r++) {
            unsigned M = r + 8u * hf, j = t * 16u + nn;
            float v = zpv[t][r] * sRed[M];
            sScr[M * 64u + j] = v;
            zbf[(e0 + M) * 64u + j] = f2bf(v);
        }
    __syncthreads();
    if (lane < 16) {
#pragma unroll
        for (unsigned hd = 0; hd < HEADS; hd++) {
            float s = 0;
            for (unsigned j = 0; j < 64; j++) s += sScr[lane * 64 + j] * wb[hd * 64u + j];
            barr[(e0 + (unsigned)lane) * 4u + hd] = s;
        }
    }
    __syncthreads();

    // ---- u = silu(chem@Wu1+bu1)@wu2+bu2 (W fragments straight from global, swizzled)
    {
        v8f acc[4];
#pragma unroll
        for (int t = 0; t < 4; t++) acc[t] = v8zero();
#pragma unroll
        for (int k0 = 0; k0 < 64; k0 += 32) {
            BfVec a = frag_ld(sChem + (k0 >> 5) * 512 + lane * 16);
#pragma unroll
            for (int t = 0; t < 4; t++)
                acc[t] = wmma_bf(a, frag_ld(wu1b + (k0 >> 5) * 2048 + t * 512 + lane * 16), acc[t]);
        }
#pragma unroll
        for (unsigned t = 0; t < 4; t++)
#pragma unroll
            for (unsigned r = 0; r < 8; r++) {
                unsigned M = r + 8u * hf, j = t * 16u + nn;
                sScr[M * 64u + j] = silu_(acc[t][r] + sBu1[j]);
            }
        __syncthreads();
    }
    float uu = 0.0f;
    if (lane < 16) {
        uu = bu2[0];
        for (int j = 0; j < 64; j++) uu += sScr[lane * 64 + j] * sWu2[j];
    }
    __syncthreads();

    // ---- x = silu(pos@Wx1+bx1)@wx2+bx2; coordinate scatter
    {
        v8f acc[4];
#pragma unroll
        for (int t = 0; t < 4; t++) acc[t] = v8zero();
#pragma unroll
        for (int k0 = 0; k0 < 64; k0 += 32) {
            BfVec a = frag_ld(sPos + (k0 >> 5) * 512 + lane * 16);
#pragma unroll
            for (int t = 0; t < 4; t++)
                acc[t] = wmma_bf(a, frag_ld(wx1b + (k0 >> 5) * 2048 + t * 512 + lane * 16), acc[t]);
        }
#pragma unroll
        for (unsigned t = 0; t < 4; t++)
#pragma unroll
            for (unsigned r = 0; r < 8; r++) {
                unsigned M = r + 8u * hf, j = t * 16u + nn;
                sScr[M * 64u + j] = silu_(acc[t][r] + sBx1[j]);
            }
        __syncthreads();
    }
    if (lane < 16) {
        float xx = bx2[0];
        for (int j = 0; j < 64; j++) xx += sScr[lane * 64 + j] * sWx2[j];
        float coef = uu * xx;
        unsigned rI = (unsigned)sRow[lane] * 3u;
#pragma unroll
        for (int k = 0; k < 3; k++) atomicAdd(&xsum[rI + k], sCd[lane * 3 + k] * coef);
        atomicAdd(&cntp[sRow[lane]], 1.0f);
    }
}

// ---------------- attention (per edge, per head), vectorized bf16 rows ----------------
__global__ void attn_kernel(const unsigned short* __restrict__ qbf,
                            const unsigned short* __restrict__ kbf,
                            const unsigned short* __restrict__ vbf,
                            const unsigned short* __restrict__ gbf,
                            const float* __restrict__ barr,
                            const int* __restrict__ klist,
                            unsigned short* __restrict__ mcat) {
    int gid = blockIdx.x * 256 + threadIdx.x;
    if (gid >= N_EDGES * HEADS) return;
    unsigned e = (unsigned)gid >> 2, hd = (unsigned)gid & 3u;
    const unsigned hoff = hd * 64u;

    float qv[64];
    {
        const v4u* q = (const v4u*)(qbf + e * 256u + hoff);
#pragma unroll
        for (int i = 0; i < 8; i++) {
            v4u x = q[i];
#pragma unroll
            for (int w = 0; w < 4; w++) {
                qv[i * 8 + w * 2]     = bflo(x[w]);
                qv[i * 8 + w * 2 + 1] = bfhi(x[w]);
            }
        }
    }
    unsigned ii[MAX_K]; float al[MAX_K];
    float mx = -1e30f;
#pragma unroll
    for (int n = 0; n < MAX_K; n++) {
        int idx = klist[e * 16u + n];
        unsigned w = (unsigned)((idx < 0) ? idx + N_EDGES : idx);  // python negative wrap
        ii[n] = w;
        __builtin_prefetch(vbf + w * 256u + hoff, 0, 1);
        float a;
        if (idx < 0) {
            a = -10000.0f;
        } else {
            const v4u* kr = (const v4u*)(kbf + w * 256u + hoff);
            float d = 0;
#pragma unroll
            for (int i = 0; i < 8; i++) {
                v4u x = kr[i];
#pragma unroll
                for (int ww = 0; ww < 4; ww++) {
                    d += qv[i * 8 + ww * 2] * bflo(x[ww]);
                    d += qv[i * 8 + ww * 2 + 1] * bfhi(x[ww]);
                }
            }
            unsigned jdx = (unsigned)klist[e * 16u + 8u + n];
            a = d * 0.125f + barr[jdx * 4u + hd];
        }
        al[n] = a; mx = fmaxf(mx, a);
    }
    float s = 0;
#pragma unroll
    for (int n = 0; n < MAX_K; n++) { al[n] = __expf(al[n] - mx); s += al[n]; }
    float inv = 1.0f / s;

    float out[64];
#pragma unroll
    for (int j = 0; j < 64; j++) out[j] = 0.0f;
#pragma unroll
    for (int n = 0; n < MAX_K; n++) {
        float w = al[n] * inv;
        const v4u* vr = (const v4u*)(vbf + ii[n] * 256u + hoff);
#pragma unroll
        for (int i = 0; i < 8; i++) {
            v4u x = vr[i];
#pragma unroll
            for (int ww = 0; ww < 4; ww++) {
                out[i * 8 + ww * 2]     += w * bflo(x[ww]);
                out[i * 8 + ww * 2 + 1] += w * bfhi(x[ww]);
            }
        }
    }
    const v4u* gr = (const v4u*)(gbf + e * 256u + hoff);
    unsigned short* mo = mcat + e * 256u + hoff;
#pragma unroll
    for (int i = 0; i < 8; i++) {
        v4u x = gr[i];
#pragma unroll
        for (int ww = 0; ww < 4; ww++) {
            mo[i * 8 + ww * 2]     = f2bf(out[i * 8 + ww * 2] * bflo(x[ww]));
            mo[i * 8 + ww * 2 + 1] = f2bf(out[i * 8 + ww * 2 + 1] * bfhi(x[ww]));
        }
    }
}

// ---------------- w_out GEMM (K=256) + atomic scatter into node sums, 4 waves ----------------
__global__ void wout_scatter(const unsigned short* __restrict__ mcat,
                             const unsigned short* __restrict__ woutb,  // swizzled
                             const float* __restrict__ b_out,
                             const int* __restrict__ edges,
                             float* __restrict__ msum) {
    __shared__ __align__(16) unsigned short sA[4][512];
    __shared__ __align__(16) unsigned short sW[2048];
    const int lane = threadIdx.x & 31;
    const int wid = threadIdx.x >> 5;
    const unsigned e0 = ((unsigned)blockIdx.x * 4u + (unsigned)wid) * 16u;
    v8f acc[4];
#pragma unroll
    for (int t = 0; t < 4; t++) acc[t] = v8zero();

    for (int k0 = 0; k0 < 256; k0 += 32) {
        stage_a_bf(sA[wid], mcat + (size_t)e0 * 256, 256u, (unsigned)k0, 256u, lane);
        if (wid == 0) {
            const v4u* src = (const v4u*)(woutb + (size_t)(k0 >> 5) * 2048);
            v4u* dst = (v4u*)sW;
            for (int i = lane; i < 256; i += 32) dst[i] = src[i];
            if (k0 + 32 < 256) __builtin_prefetch(woutb + (size_t)((k0 + 32) >> 5) * 2048, 0, 1);
        }
        __syncthreads();
        BfVec a = frag_ld(sA[wid] + lane * 16);
#pragma unroll
        for (int t = 0; t < 4; t++)
            acc[t] = wmma_bf(a, frag_ld(sW + t * 512 + lane * 16), acc[t]);
        __syncthreads();
    }
    unsigned hf = (unsigned)lane >> 4, nn = (unsigned)lane & 15u;
#pragma unroll
    for (unsigned t = 0; t < 4; t++)
#pragma unroll
        for (unsigned r = 0; r < 8; r++) {
            unsigned M = r + 8u * hf, j = t * 16u + nn;
            unsigned row = (unsigned)edges[e0 + M];
            atomicAdd(&msum[row * 64u + j], acc[t][r] + b_out[j]);
        }
}

// ---------------- node finalize ----------------
__global__ void node_fin(const float* __restrict__ coord, const float* __restrict__ init_coord,
                         const float* __restrict__ xsum, const float* __restrict__ cntp,
                         float* __restrict__ msum, float* __restrict__ outCoord) {
    int i = blockIdx.x * 128 + threadIdx.x;
    if (i >= N_NODES) return;
    float inv = 1.0f / fmaxf(cntp[i], 1.0f);
#pragma unroll
    for (int k = 0; k < 3; k++)
        outCoord[i * 3 + k] = 0.2f * init_coord[i * 3 + k] + 0.8f * coord[i * 3 + k] + xsum[i * 3 + k] * inv;
    for (unsigned j = 0; j < 64; j++) msum[(unsigned)i * 64u + j] *= inv;
}

// =====================================================================================
extern "C" void kernel_launch(void* const* d_in, const int* in_sizes, int n_in,
                              void* d_out, int out_size, void* d_ws, size_t ws_size,
                              hipStream_t stream) {
    const float* h        = (const float*)d_in[0];
    const float* coord    = (const float*)d_in[1];
    const int*   edges    = (const int*)d_in[2];
    const float* nvecs    = (const float*)d_in[3];
    const float* edge_attr= (const float*)d_in[4];
    const float* node_attr= (const float*)d_in[5];
    const float* init_coord=(const float*)d_in[6];
    const int*   klist    = (const int*)d_in[7];
    const float* w_chem1  = (const float*)d_in[8];
    const float* b_chem1  = (const float*)d_in[9];
    const float* w_chem2  = (const float*)d_in[10];
    const float* b_chem2  = (const float*)d_in[11];
    const float* w_pos1   = (const float*)d_in[12];
    const float* b_pos1   = (const float*)d_in[13];
    const float* w_pos2   = (const float*)d_in[14];
    const float* b_pos2   = (const float*)d_in[15];
    const float* w_sh     = (const float*)d_in[16];
    const float* b_sh     = (const float*)d_in[17];
    const float* w_att    = (const float*)d_in[18];
    const float* b_att    = (const float*)d_in[19];
    const float* wq       = (const float*)d_in[20];
    const float* wk       = (const float*)d_in[21];
    const float* wv       = (const float*)d_in[22];
    const float* wb       = (const float*)d_in[23];
    const float* wg       = (const float*)d_in[24];
    const float* bg       = (const float*)d_in[25];
    const float* w_out    = (const float*)d_in[26];
    const float* b_out    = (const float*)d_in[27];
    const float* wu1      = (const float*)d_in[28];
    const float* bu1      = (const float*)d_in[29];
    const float* wu2      = (const float*)d_in[30];
    const float* bu2      = (const float*)d_in[31];
    const float* wx1      = (const float*)d_in[32];
    const float* bx1      = (const float*)d_in[33];
    const float* wx2      = (const float*)d_in[34];
    const float* bx2      = (const float*)d_in[35];
    const float* wh1      = (const float*)d_in[36];
    const float* bh1      = (const float*)d_in[37];
    const float* wh2      = (const float*)d_in[38];
    const float* bh2      = (const float*)d_in[39];

    float* outp = (float*)d_out;           // [h_new (10000x64) | coord_new (10000x3)]
    float* outCoord = outp + (size_t)N_NODES * 64;

    // ---------------- workspace layout ----------------
    size_t off = 0;
    auto take = [&](size_t bytes) -> size_t { size_t o = off; off += (bytes + 255) & ~(size_t)255; return o; };
    char* ws = (char*)d_ws;
#define WSP(type, name, count) type* name = (type*)(ws + take((size_t)(count) * sizeof(type)));
    WSP(unsigned short, wc1aS, 4096)   // h[row] block
    WSP(unsigned short, wc1bS, 4096)   // h[col] block
    WSP(unsigned short, wc1cS, 4096)   // node_attr[row] block
    WSP(unsigned short, wc1dS, 4096)   // node_attr[col] block
    WSP(unsigned short, wc1eS, 2048)   // edge_attr block (K=16 padded)
    WSP(unsigned short, wc2S, 4096)
    WSP(unsigned short, wshS, 4096)
    WSP(unsigned short, wp1eS, 2048)
    WSP(unsigned short, wp2S, 4096)
    WSP(unsigned short, wu1S, 4096)
    WSP(unsigned short, wx1S, 4096)
    WSP(unsigned short, wqS, 4 * 4096)
    WSP(unsigned short, wkS, 4 * 4096)
    WSP(unsigned short, wvS, 4 * 4096)
    WSP(unsigned short, wgS, 4 * 4096)
    WSP(unsigned short, woutS, 16384)
    WSP(unsigned short, wh1aS, 4096)
    WSP(unsigned short, wh1bS, 4096)
    WSP(unsigned short, wh1cS, 4096)
    WSP(unsigned short, wh2S, 4096)
    WSP(float, P1, N_NODES * 64)
    WSP(float, P2, N_NODES * 64)
    WSP(float, zreg, N_NODES * 68)          // xsum(3N) | cnt(N) | msum(64N)
    float* xsum = zreg;
    float* cntp = zreg + N_NODES * 3;
    float* msum = zreg + N_NODES * 4;
    WSP(float, barr, (size_t)N_EDGES * 4)
    WSP(float, hacc, N_NODES * 64)
    WSP(unsigned short, hmidb, N_NODES * 64)
    WSP(unsigned short, zb, (size_t)N_EDGES * 64)
    WSP(unsigned short, qb, (size_t)N_EDGES * 256)
    WSP(unsigned short, kb, (size_t)N_EDGES * 256)
    WSP(unsigned short, vb, (size_t)N_EDGES * 256)
    WSP(unsigned short, gb, (size_t)N_EDGES * 256)
    WSP(unsigned short, mc, (size_t)N_EDGES * 256)
#undef WSP
    (void)ws_size; (void)in_sizes; (void)n_in; (void)out_size;

    // ---------------- weight convert + swizzle ----------------
#define CVTS(src, dst, Ksrc, Kpad) \
    cvt_swz_kernel<<<((Kpad) * 64 + 255) / 256, 256, 0, stream>>>(src, dst, Ksrc, Kpad)
    CVTS(w_chem1,            wc1aS, 64, 64);
    CVTS(w_chem1 + 64 * 64,  wc1bS, 64, 64);
    CVTS(w_chem1 + 128 * 64, wc1cS, 64, 64);
    CVTS(w_chem1 + 192 * 64, wc1dS, 64, 64);
    CVTS(w_chem1 + 256 * 64, wc1eS, 16, 32);
    CVTS(w_chem2, wc2S, 64, 64);
    CVTS(w_sh,    wshS, 64, 64);
    CVTS(w_pos1 + 20 * 64, wp1eS, 16, 32);
    CVTS(w_pos2,  wp2S, 64, 64);
    CVTS(wu1,     wu1S, 64, 64);
    CVTS(wx1,     wx1S, 64, 64);
    for (int hd = 0; hd < HEADS; hd++) {
        CVTS(wq + hd * 4096, wqS + hd * 4096, 64, 64);
        CVTS(wk + hd * 4096, wkS + hd * 4096, 64, 64);
        CVTS(wv + hd * 4096, wvS + hd * 4096, 64, 64);
        CVTS(wg + hd * 4096, wgS + hd * 4096, 64, 64);
    }
    CVTS(w_out, woutS, 256, 256);
    CVTS(wh1,            wh1aS, 64, 64);
    CVTS(wh1 + 64 * 64,  wh1bS, 64, 64);
    CVTS(wh1 + 128 * 64, wh1cS, 64, 64);
    CVTS(wh2, wh2S, 64, 64);
#undef CVTS
    zero_kernel<<<(N_NODES * 68 + 255) / 256, 256, 0, stream>>>(zreg, N_NODES * 68);

    const int NT = N_NODES / 16;   // 625 tiles
    const int ET = N_EDGES / 16;   // 10000 tiles
    const int NB = (NT + 3) / 4;   // 157 blocks
    const int EB = ET / 4;         // 2500 blocks

    // ---- node projections P1/P2 (factored chem layer-1) ----
    gemm64<0, false, false, false, false, false, true><<<NB, 128, 0, stream>>>(
        h, 64, 64, wc1aS, nullptr, nullptr, nullptr, P1, nullptr, 64, 0, NT);
    gemm64<0, true, false, false, false, false, true><<<NB, 128, 0, stream>>>(
        node_attr, 64, 64, wc1cS, nullptr, P1, nullptr, P1, nullptr, 64, 0, NT);
    gemm64<0, false, false, false, false, false, true><<<NB, 128, 0, stream>>>(
        h, 64, 64, wc1bS, nullptr, nullptr, nullptr, P2, nullptr, 64, 0, NT);
    gemm64<0, true, false, false, false, false, true><<<NB, 128, 0, stream>>>(
        node_attr, 64, 64, wc1dS, nullptr, P2, nullptr, P2, nullptr, 64, 0, NT);

    // ---- fused edge pipeline ----
    edge_fused<<<ET, 32, 0, stream>>>(
        coord, edges, nvecs, edge_attr, P1, P2,
        wc2S, wshS, wp2S, wu1S, wx1S,
        wc1eS, wp1eS, w_pos1,
        b_chem1, b_chem2, b_sh, b_pos1, b_pos2, bu1, bx1,
        w_att, b_att, wu2, bu2, wx2, bx2, wb,
        zb, barr, xsum, cntp);

    // ---- per-head q/k/v/g projections ----
    for (int hd = 0; hd < HEADS; hd++) {
        gemm64<0, false, true, false, false, true, false><<<EB, 128, 0, stream>>>(
            zb, 64, 64, wqS + hd * 4096, nullptr, nullptr, nullptr, nullptr, qb, 256, hd * 64, ET);
        gemm64<0, false, true, false, false, true, false><<<EB, 128, 0, stream>>>(
            zb, 64, 64, wkS + hd * 4096, nullptr, nullptr, nullptr, nullptr, kb, 256, hd * 64, ET);
        gemm64<0, false, true, false, false, true, false><<<EB, 128, 0, stream>>>(
            zb, 64, 64, wvS + hd * 4096, nullptr, nullptr, nullptr, nullptr, vb, 256, hd * 64, ET);
        gemm64<2, false, true, true, false, true, false><<<EB, 128, 0, stream>>>(
            zb, 64, 64, wgS + hd * 4096, bg + hd * 64, nullptr, nullptr, nullptr, gb, 256, hd * 64, ET);
    }

    // ---- attention ----
    attn_kernel<<<(N_EDGES * HEADS + 255) / 256, 256, 0, stream>>>(qb, kb, vb, gb, barr, klist, mc);

    // ---- m = mcat @ w_out + b_out, scattered into node sums ----
    wout_scatter<<<EB, 128, 0, stream>>>(mc, woutS, b_out, edges, msum);

    // ---- node finalize: coord output + segment means ----
    node_fin<<<(N_NODES + 127) / 128, 128, 0, stream>>>(coord, init_coord, xsum, cntp, msum, outCoord);

    // ---- h update MLP ----
    gemm64<0, false, false, false, false, false, true><<<NB, 128, 0, stream>>>(
        h, 64, 64, wh1aS, nullptr, nullptr, nullptr, hacc, nullptr, 64, 0, NT);
    gemm64<0, true, false, false, false, false, true><<<NB, 128, 0, stream>>>(
        node_attr, 64, 64, wh1bS, nullptr, hacc, nullptr, hacc, nullptr, 64, 0, NT);
    gemm64<1, true, false, true, false, true, false><<<NB, 128, 0, stream>>>(
        msum, 64, 64, wh1cS, bh1, hacc, nullptr, nullptr, hmidb, 64, 0, NT);
    gemm64<0, false, true, true, true, false, true><<<NB, 128, 0, stream>>>(
        hmidb, 64, 64, wh2S, bh2, nullptr, h, outp, nullptr, 64, 0, NT);
}
